// XenoglauxModel_90675349553884
// MI455X (gfx1250) — matrix operand order repaired and checked by reading, not compiled, not run on hardware
//
#include <hip/hip_runtime.h>
#include <hip/hip_bf16.h>
#include <math.h>

// ---------------- model constants ----------------
#define LAYERS 12
#define DMODEL 768
#define NHEAD  12
#define DKH    64
#define FFDIM  3072
#define VOCAB  32000
#define BS     2
#define SEQ    1024
#define BT     (BS * SEQ)
#define LN_EPS 1e-5f

typedef unsigned short u16;
typedef __attribute__((ext_vector_type(16))) __bf16 v16bf;
typedef __attribute__((ext_vector_type(8)))  __bf16 v8bf;
typedef __attribute__((ext_vector_type(8)))  float  v8f;
typedef __attribute__((ext_vector_type(8)))  u16    u16x8;
typedef __attribute__((ext_vector_type(4)))  unsigned int v4u;
typedef __attribute__((ext_vector_type(4)))  int    v4i;
typedef __attribute__((ext_vector_type(8)))  int    v8i;

#ifndef __has_builtin
#define __has_builtin(x) 0
#endif
#if defined(__HIP_DEVICE_COMPILE__) && __has_builtin(__builtin_amdgcn_tensor_load_to_lds)
#define USE_TDM 1
#else
#define USE_TDM 0
#endif

#if defined(__HIP_DEVICE_COMPILE__) && !USE_TDM
#warning "gfx1250 TDM builtin unavailable: k_gemm using cooperative-load fallback"
#endif

// ---------------- helpers ----------------
__device__ __forceinline__ u16 f2bf(float f) {
  unsigned u = __builtin_bit_cast(unsigned, f);
  u += 0x7FFFu + ((u >> 16) & 1u);            // round-to-nearest-even
  return (u16)(u >> 16);
}
__device__ __forceinline__ __bf16 u2b(u16 u) {
  return __builtin_bit_cast(__bf16, u);
}

// A-fragment: 16x32 bf16 from row-major source, stride lda (elements), Ap at (m0,k0).
// ISA: lanes0-15 row M=lane; halves j=0..7 -> K=grp*8+j, j=8..15 -> K=16+grp*8+(j-8).
__device__ __forceinline__ v16bf frag_a(const u16* Ap, int lda, int lane) {
  int grp = lane >> 4, row = lane & 15;
  const u16* r = Ap + (size_t)row * lda + grp * 8;
  u16x8 lo = *(const u16x8*)r;
  u16x8 hi = *(const u16x8*)(r + 16);
  v8bf bl = __builtin_bit_cast(v8bf, lo);
  v8bf bh = __builtin_bit_cast(v8bf, hi);
  return __builtin_shufflevector(bl, bh, 0,1,2,3,4,5,6,7,8,9,10,11,12,13,14,15);
}

// B-fragment from a TRANSPOSED source S (row-major [N][K], stride lds_):
// B[k][n] = S[n][k]; lane holds col n=lane&15, j=0..7 -> k=(lane>>4)*8+j, j=8..15 -> k=16+(lane>>4)*8+(j-8).
__device__ __forceinline__ v16bf frag_bT(const u16* Sp, int lds_, int lane) {
  int grp = lane >> 4, col = lane & 15;
  const u16* r = Sp + (size_t)col * lds_ + grp * 8;
  u16x8 lo = *(const u16x8*)r;
  u16x8 hi = *(const u16x8*)(r + 16);
  v8bf bl = __builtin_bit_cast(v8bf, lo);
  v8bf bh = __builtin_bit_cast(v8bf, hi);
  return __builtin_shufflevector(bl, bh, 0,1,2,3,4,5,6,7,8,9,10,11,12,13,14,15);
}

#if USE_TDM
// TDM: 2D tile load (64 rows x tile_w cols of bf16) from row-major [M][K] into LDS.
// D# packing per CDNA5 ISA ch.8 (group0 128b, group1 256b, groups2/3 zero for 2D).
__device__ __forceinline__ void tdm_load_a(const u16* gptr, unsigned lds_off,
                                           int K, int M, unsigned tile_w) {
  unsigned long long ga = (unsigned long long)gptr;
  v4u g0;
  g0[0] = 1u;                                          // count=1, user descriptor
  g0[1] = lds_off;                                     // lds_addr (bytes)
  g0[2] = (unsigned)ga;                                // global_addr[31:0]
  g0[3] = (unsigned)((ga >> 32) & 0x1FFFFFFu) | (2u << 30);  // addr[56:32] | type=2
  v8i g1;
  g1[0] = (int)(1u << 16);                             // wg_mask=0, data_size=1 (2B)
  g1[1] = (int)(((unsigned)K & 0xFFFFu) << 16);        // tensor_dim0[15:0] @bits63:48
  g1[2] = (int)(((unsigned)K >> 16) | (((unsigned)M & 0xFFFFu) << 16)); // dim0 hi | dim1 lo
  g1[3] = (int)(((unsigned)M >> 16) | (tile_w << 16)); // dim1 hi | tile_dim0
  g1[4] = 64;                                          // tile_dim1=64, tile_dim2=0
  g1[5] = K;                                           // tensor_dim0_stride[31:0]
  g1[6] = 0;                                           // stride hi | dim1_stride lo
  g1[7] = 0;
  v4i z4 = {0, 0, 0, 0};
#if defined(__clang_major__) && __clang_major__ >= 23
  v8i z8 = {0, 0, 0, 0, 0, 0, 0, 0};
  __builtin_amdgcn_tensor_load_to_lds(g0, g1, z4, z4, z8, 0);
#else
  __builtin_amdgcn_tensor_load_to_lds(g0, g1, z4, z4, 0);
#endif
}
#endif

// ---------------- elementwise / conversion kernels ----------------
__global__ void k_f32_to_bf16(const float* __restrict__ src, u16* __restrict__ dst, long long n) {
  long long i = (long long)blockIdx.x * blockDim.x + threadIdx.x;
  long long st = (long long)gridDim.x * blockDim.x;
  for (; i < n; i += st) dst[i] = f2bf(src[i]);
}

// per-layer transpose-convert: dst[l][n][k] = bf16(src[l][k][n])   ([K][N] -> [N][K])
__global__ void k_convT(const float* __restrict__ src, u16* __restrict__ dst,
                        int Kd, int Nd, int layers) {
  long long tot = (long long)layers * Kd * Nd;
  long long i = (long long)blockIdx.x * blockDim.x + threadIdx.x;
  long long st = (long long)gridDim.x * blockDim.x;
  for (; i < tot; i += st) {
    long long l = i / ((long long)Kd * Nd);
    long long rem = i % ((long long)Kd * Nd);
    long long nn = rem / Kd, kk = rem % Kd;
    dst[i] = f2bf(src[(size_t)l * Kd * Nd + (size_t)kk * Nd + nn]);
  }
}

__global__ void k_embed(const int* __restrict__ tok, const float* __restrict__ emb,
                        const float* __restrict__ pos, float* __restrict__ x) {
  long long n = (long long)BT * DMODEL;
  long long i = (long long)blockIdx.x * blockDim.x + threadIdx.x;
  long long st = (long long)gridDim.x * blockDim.x;
  for (; i < n; i += st) {
    long long bt = i / DMODEL, d = i % DMODEL;
    long long t = bt % SEQ;
    x[i] = emb[(size_t)tok[bt] * DMODEL + d] + pos[(size_t)t * DMODEL + d];
  }
}

// qkv [BT,3D] fp32 -> q,k bf16 [B,H,T,DK]; v bf16 TRANSPOSED [B,H,DK,T]
__global__ void k_qkv_split(const float* __restrict__ qkv, u16* __restrict__ q,
                            u16* __restrict__ k, u16* __restrict__ vT) {
  long long n = (long long)BS * NHEAD * SEQ * DKH;
  long long i = (long long)blockIdx.x * blockDim.x + threadIdx.x;
  long long st = (long long)gridDim.x * blockDim.x;
  for (; i < n; i += st) {
    long long d = i % DKH;
    long long t = (i / DKH) % SEQ;
    long long h = (i / ((long long)DKH * SEQ)) % NHEAD;
    long long b = i / ((long long)DKH * SEQ * NHEAD);
    size_t src = ((size_t)(b * SEQ + t)) * (3 * DMODEL) + h * DKH + d;
    q[i] = f2bf(qkv[src]);
    k[i] = f2bf(qkv[src + DMODEL]);
    vT[((size_t)(b * NHEAD + h) * DKH + d) * SEQ + t] = f2bf(qkv[src + 2 * DMODEL]);
  }
}

// LayerNorm over D=768, one WG per row, bf16 output
__global__ __launch_bounds__(256) void k_ln(const float* __restrict__ x,
                                            const float* __restrict__ w,
                                            const float* __restrict__ bb,
                                            u16* __restrict__ o) {
  __shared__ float s1[256], s2[256];
  int row = blockIdx.x, tid = threadIdx.x;
  const float* r = x + (size_t)row * DMODEL;
  float a = 0.f, qq = 0.f;
  for (int j = tid; j < DMODEL; j += 256) { float v = r[j]; a += v; qq += v * v; }
  s1[tid] = a; s2[tid] = qq; __syncthreads();
  for (int st = 128; st > 0; st >>= 1) {
    if (tid < st) { s1[tid] += s1[tid + st]; s2[tid] += s2[tid + st]; }
    __syncthreads();
  }
  float mean = s1[0] * (1.0f / DMODEL);
  float var  = s2[0] * (1.0f / DMODEL) - mean * mean;
  float inv  = rsqrtf(var + LN_EPS);
  for (int j = tid; j < DMODEL; j += 256)
    o[(size_t)row * DMODEL + j] = f2bf((r[j] - mean) * inv * w[j] + bb[j]);
}

// causal row softmax over T=1024, bf16 output
__global__ __launch_bounds__(256) void k_softmax(const float* __restrict__ s, u16* __restrict__ p) {
  __shared__ float red[256];
  int row = blockIdx.x, tid = threadIdx.x;
  const float* r = s + (size_t)row * SEQ;
  float lm = -__builtin_inff();
  for (int j = tid; j < SEQ; j += 256) lm = fmaxf(lm, r[j]);
  red[tid] = lm; __syncthreads();
  for (int st = 128; st > 0; st >>= 1) { if (tid < st) red[tid] = fmaxf(red[tid], red[tid + st]); __syncthreads(); }
  float m = red[0]; __syncthreads();
  float ls = 0.f;
  for (int j = tid; j < SEQ; j += 256) ls += expf(r[j] - m);
  red[tid] = ls; __syncthreads();
  for (int st = 128; st > 0; st >>= 1) { if (tid < st) red[tid] += red[tid + st]; __syncthreads(); }
  float inv = 1.0f / red[0];
  for (int j = tid; j < SEQ; j += 256)
    p[(size_t)row * SEQ + j] = f2bf(expf(r[j] - m) * inv);
}

// ---------------- generic bf16 WMMA GEMM ----------------
// out[M,N] = A[M,K](bf16) @ W (given TRANSPOSED: WT[N][K] bf16) + bias (+GELU) (+resid)
// A tile (64x64) staged in LDS (TDM double-buffered); 8 WMMAs/wave per barrier pair.
template <int GELU, int HAS_BIAS, int HAS_RESID>
__global__ __launch_bounds__(256) void k_gemm(const u16* __restrict__ A, const u16* __restrict__ WT,
                                              const float* __restrict__ bias,
                                              const float* __restrict__ resid,
                                              float* __restrict__ out,
                                              int M, int N, int K) {
  __shared__ alignas(16) u16 lA[2][64 * 64];
  int tid = threadIdx.x, lane = tid & 31, wid = tid >> 5;
  int wm = wid & 3, wn = wid >> 2;
  int m0 = blockIdx.y * 64, n0 = blockIdx.x * 128;
  v8f acc[4] = {};
  int steps = K / 64;

#if USE_TDM
  if (wid == 0)
    tdm_load_a(A + (size_t)m0 * K, (unsigned)(unsigned long long)&lA[0][0], K, M, 64u);
  for (int i = 0; i < steps; ++i) {
    int k0 = i * 64;
    __syncthreads();                       // prior reads of lA[(i+1)&1] complete
    if (wid == 0) {
      if (i + 1 < steps) {
        tdm_load_a(A + (size_t)m0 * K + (k0 + 64),
                   (unsigned)(unsigned long long)&lA[(i + 1) & 1][0], K, M, 64u);
        __builtin_amdgcn_s_wait_tensorcnt(1);   // tile i done (in-order), tile i+1 in flight
      } else {
        __builtin_amdgcn_s_wait_tensorcnt(0);
      }
    }
    __syncthreads();                       // lA[i&1] ready for all waves
    const u16* abase = &lA[i & 1][(wm * 16) * 64];
    v16bf a0 = frag_a(abase, 64, lane);
    v16bf a1 = frag_a(abase + 32, 64, lane);
    const u16* wbase = WT + (size_t)(n0 + wn * 64) * K + k0;
#pragma unroll
    for (int nt = 0; nt < 4; ++nt) {
      v16bf b0 = frag_bT(wbase + (size_t)nt * 16 * K, K, lane);
      acc[nt] = __builtin_amdgcn_wmma_f32_16x16x32_bf16(false, a0, false, b0,
                                                        (short)0, acc[nt], false, false);
      v16bf b1 = frag_bT(wbase + (size_t)nt * 16 * K + 32, K, lane);
      acc[nt] = __builtin_amdgcn_wmma_f32_16x16x32_bf16(false, a1, false, b1,
                                                        (short)0, acc[nt], false, false);
    }
  }
#else
  for (int i = 0; i < steps; ++i) {
    int k0 = i * 64;
#pragma unroll
    for (int s = tid; s < 512; s += 256) {  // stage 64x64: 8 bf16 x2 per thread, 16B aligned
      int row = s >> 3, col = (s & 7) * 8;
      u16x8 v = *(const u16x8*)(A + (size_t)(m0 + row) * K + k0 + col);
      *(u16x8*)(&lA[0][row * 64 + col]) = v;
    }
    __syncthreads();
    const u16* abase = &lA[0][(wm * 16) * 64];
    v16bf a0 = frag_a(abase, 64, lane);
    v16bf a1 = frag_a(abase + 32, 64, lane);
    const u16* wbase = WT + (size_t)(n0 + wn * 64) * K + k0;
#pragma unroll
    for (int nt = 0; nt < 4; ++nt) {
      v16bf b0 = frag_bT(wbase + (size_t)nt * 16 * K, K, lane);
      acc[nt] = __builtin_amdgcn_wmma_f32_16x16x32_bf16(false, a0, false, b0,
                                                        (short)0, acc[nt], false, false);
      v16bf b1 = frag_bT(wbase + (size_t)nt * 16 * K + 32, K, lane);
      acc[nt] = __builtin_amdgcn_wmma_f32_16x16x32_bf16(false, a1, false, b1,
                                                        (short)0, acc[nt], false, false);
    }
    __syncthreads();
  }
#endif

  int grp = lane >> 4, cl = lane & 15;
#pragma unroll
  for (int nt = 0; nt < 4; ++nt) {
    int c = n0 + wn * 64 + nt * 16 + cl;
    float bv = HAS_BIAS ? bias[c] : 0.0f;
#pragma unroll
    for (int i = 0; i < 8; ++i) {
      int r = m0 + wm * 16 + i + 8 * grp;
      float v = acc[nt][i] + bv;
      if (GELU) v = 0.5f * v * (1.0f + erff(v * 0.70710678118654752f));
      if (HAS_RESID) v += resid[(size_t)r * N + c];
      out[(size_t)r * N + c] = v;
    }
  }
}

// ---------------- attention: S = scale * Q K^T with causal mask ----------------
__global__ __launch_bounds__(256) void k_scores(const u16* __restrict__ q,
                                                const u16* __restrict__ kmat,
                                                float* __restrict__ s, float scale) {
  int lane = threadIdx.x & 31, wid = threadIdx.x >> 5;
  int bh = blockIdx.x, qt = blockIdx.y, kt = blockIdx.z * 8 + wid;
  int grp = lane >> 4, cl = lane & 15;
  float* srow = s + (size_t)bh * SEQ * SEQ;

  if (kt > qt) {  // fully above diagonal: no compute
#pragma unroll
    for (int i = 0; i < 8; ++i) {
      int qi = qt * 16 + i + 8 * grp, ki = kt * 16 + cl;
      srow[(size_t)qi * SEQ + ki] = -__builtin_inff();
    }
    return;
  }
  const u16* qb = q    + ((size_t)bh * SEQ + qt * 16) * DKH;
  const u16* kb = kmat + ((size_t)bh * SEQ + kt * 16) * DKH;
  v8f acc = {};
#pragma unroll
  for (int k0 = 0; k0 < DKH; k0 += 32) {
    v16bf a = frag_a(qb + k0, DKH, lane);
    v16bf b = frag_bT(kb + k0, DKH, lane);   // K^T: B[k][n] = K[n][k]
    acc = __builtin_amdgcn_wmma_f32_16x16x32_bf16(false, a, false, b,
                                                  (short)0, acc, false, false);
  }
#pragma unroll
  for (int i = 0; i < 8; ++i) {
    int qi = qt * 16 + i + 8 * grp, ki = kt * 16 + cl;
    srow[(size_t)qi * SEQ + ki] = (ki <= qi) ? acc[i] * scale : -__builtin_inff();
  }
}

// ---------------- attention: Y = P V, V given transposed [B,H,DK,T] ----------------
__global__ __launch_bounds__(256) void k_pv(const u16* __restrict__ p,
                                            const u16* __restrict__ vT,
                                            float* __restrict__ y) {
  int lane = threadIdx.x & 31, wid = threadIdx.x >> 5;
  int wm = wid & 1, wn = wid >> 1;
  int bh = blockIdx.x, m0 = blockIdx.y * 32;
  int b = bh / NHEAD, h = bh % NHEAD;
  const u16* pb = p  + (size_t)bh * SEQ * SEQ + (size_t)(m0 + wm * 16) * SEQ;
  const u16* vb = vT + ((size_t)bh * DKH + wn * 16) * SEQ;
  v8f acc = {};
  int kmax = m0 + 32;                  // causal: P is zero beyond the diagonal
  for (int k0 = 0; k0 < kmax; k0 += 32) {
    v16bf a  = frag_a(pb + k0, SEQ, lane);
    v16bf bb = frag_bT(vb + k0, SEQ, lane);  // B[k][n] = vT[n][k]
    acc = __builtin_amdgcn_wmma_f32_16x16x32_bf16(false, a, false, bb,
                                                  (short)0, acc, false, false);
  }
  int grp = lane >> 4, cl = lane & 15;
#pragma unroll
  for (int i = 0; i < 8; ++i) {
    int t = m0 + wm * 16 + i + 8 * grp;
    int c = h * DKH + wn * 16 + cl;
    y[((size_t)b * SEQ + t) * DMODEL + c] = acc[i];
  }
}

// ---------------- host orchestration ----------------
extern "C" void kernel_launch(void* const* d_in, const int* in_sizes, int n_in,
                              void* d_out, int out_size, void* d_ws, size_t ws_size,
                              hipStream_t stream) {
  (void)in_sizes; (void)n_in; (void)out_size; (void)ws_size;
  const int*   tokens = (const int*)  d_in[0];
  const float* emb    = (const float*)d_in[1];
  const float* pos    = (const float*)d_in[2];
  const float* ln1w   = (const float*)d_in[3];
  const float* ln1b   = (const float*)d_in[4];
  const float* qkvw   = (const float*)d_in[5];
  const float* qkvb   = (const float*)d_in[6];
  const float* projw  = (const float*)d_in[7];
  const float* projb  = (const float*)d_in[8];
  const float* ln2w   = (const float*)d_in[9];
  const float* ln2b   = (const float*)d_in[10];
  const float* ff1w   = (const float*)d_in[11];
  const float* ff1b   = (const float*)d_in[12];
  const float* ff2w   = (const float*)d_in[13];
  const float* ff2b   = (const float*)d_in[14];
  const float* lnfw   = (const float*)d_in[15];
  const float* lnfb   = (const float*)d_in[16];
  float* out = (float*)d_out;

  char* ws = (char*)d_ws;
  size_t off = 0;
  auto alloc = [&](size_t bytes) -> char* {
    char* p = ws + off;
    off = (off + bytes + 255) & ~(size_t)255;
    return p;
  };
  // weights, pre-transposed to [N][K] bf16 (emb is already W^T for the tied head)
  u16*   qkvwT  = (u16*)  alloc((size_t)LAYERS * DMODEL * 3 * DMODEL * 2);
  u16*   projwT = (u16*)  alloc((size_t)LAYERS * DMODEL * DMODEL * 2);
  u16*   ff1wT  = (u16*)  alloc((size_t)LAYERS * DMODEL * FFDIM * 2);
  u16*   ff2wT  = (u16*)  alloc((size_t)LAYERS * FFDIM * DMODEL * 2);
  u16*   embWT  = (u16*)  alloc((size_t)VOCAB * DMODEL * 2);
  float* x0     = (float*)alloc((size_t)BT * DMODEL * 4);
  float* x1     = (float*)alloc((size_t)BT * DMODEL * 4);
  u16*   h_bf   = (u16*)  alloc((size_t)BT * DMODEL * 2);
  float* qkvbuf = (float*)alloc((size_t)BT * 3 * DMODEL * 4);
  u16*   q_bf   = (u16*)  alloc((size_t)BS * NHEAD * SEQ * DKH * 2);
  u16*   k_bf   = (u16*)  alloc((size_t)BS * NHEAD * SEQ * DKH * 2);
  u16*   vT_bf  = (u16*)  alloc((size_t)BS * NHEAD * SEQ * DKH * 2);
  float* scores = (float*)alloc((size_t)BS * NHEAD * SEQ * SEQ * 4);
  u16*   p_bf   = (u16*)  alloc((size_t)BS * NHEAD * SEQ * SEQ * 2);
  float* yb     = (float*)alloc((size_t)BT * DMODEL * 4);
  u16*   y_bf   = (u16*)  alloc((size_t)BT * DMODEL * 2);
  float* ffh    = (float*)alloc((size_t)BT * FFDIM * 4);
  u16*   ffh_bf = (u16*)  alloc((size_t)BT * FFDIM * 2);

  dim3 blk(256);
  int cg = 4096;
  // weight conversions (fp32 -> bf16, transposed where needed)
  k_convT<<<cg, blk, 0, stream>>>(qkvw,  qkvwT,  DMODEL, 3 * DMODEL, LAYERS);
  k_convT<<<cg, blk, 0, stream>>>(projw, projwT, DMODEL, DMODEL,     LAYERS);
  k_convT<<<cg, blk, 0, stream>>>(ff1w,  ff1wT,  DMODEL, FFDIM,      LAYERS);
  k_convT<<<cg, blk, 0, stream>>>(ff2w,  ff2wT,  FFDIM,  DMODEL,     LAYERS);
  k_f32_to_bf16<<<cg, blk, 0, stream>>>(emb, embWT, (long long)VOCAB * DMODEL);

  k_embed<<<cg, blk, 0, stream>>>(tokens, emb, pos, x0);

  for (int l = 0; l < LAYERS; ++l) {
    const u16* wq = qkvwT  + (size_t)l * DMODEL * 3 * DMODEL;
    const u16* wp = projwT + (size_t)l * DMODEL * DMODEL;
    const u16* w1 = ff1wT  + (size_t)l * DMODEL * FFDIM;
    const u16* w2 = ff2wT  + (size_t)l * FFDIM * DMODEL;

    k_ln<<<BT, blk, 0, stream>>>(x0, ln1w + (size_t)l * DMODEL, ln1b + (size_t)l * DMODEL, h_bf);
    k_gemm<0, 1, 0><<<dim3(3 * DMODEL / 128, BT / 64), blk, 0, stream>>>(
        h_bf, wq, qkvb + (size_t)l * 3 * DMODEL, nullptr, qkvbuf, BT, 3 * DMODEL, DMODEL);
    k_qkv_split<<<cg, blk, 0, stream>>>(qkvbuf, q_bf, k_bf, vT_bf);

    k_scores<<<dim3(BS * NHEAD, SEQ / 16, 8), blk, 0, stream>>>(q_bf, k_bf, scores, 0.125f);
    k_softmax<<<BS * NHEAD * SEQ, blk, 0, stream>>>(scores, p_bf);
    k_pv<<<dim3(BS * NHEAD, SEQ / 32), blk, 0, stream>>>(p_bf, vT_bf, yb);

    k_f32_to_bf16<<<cg, blk, 0, stream>>>(yb, y_bf, (long long)BT * DMODEL);
    k_gemm<0, 1, 1><<<dim3(DMODEL / 128, BT / 64), blk, 0, stream>>>(
        y_bf, wp, projb + (size_t)l * DMODEL, x0, x1, BT, DMODEL, DMODEL);

    k_ln<<<BT, blk, 0, stream>>>(x1, ln2w + (size_t)l * DMODEL, ln2b + (size_t)l * DMODEL, h_bf);
    k_gemm<1, 1, 0><<<dim3(FFDIM / 128, BT / 64), blk, 0, stream>>>(
        h_bf, w1, ff1b + (size_t)l * FFDIM, nullptr, ffh, BT, FFDIM, DMODEL);
    k_f32_to_bf16<<<cg, blk, 0, stream>>>(ffh, ffh_bf, (long long)BT * FFDIM);
    k_gemm<0, 1, 1><<<dim3(DMODEL / 128, BT / 64), blk, 0, stream>>>(
        ffh_bf, w2, ff2b + (size_t)l * DMODEL, x1, x0, BT, DMODEL, FFDIM);
  }

  k_ln<<<BT, blk, 0, stream>>>(x0, lnfw, lnfb, h_bf);
  k_gemm<0, 0, 0><<<dim3(VOCAB / 128, BT / 64), blk, 0, stream>>>(
      h_bf, embWT, nullptr, nullptr, out, BT, VOCAB, DMODEL);
}
